// MLA_39436389712281
// MI455X (gfx1250) — matrix operand order repaired and checked by reading, not compiled, not run on hardware
//
#include <hip/hip_runtime.h>
#include <hip/hip_bf16.h>

// ---------------------------------------------------------------------------
// MLA pipeline for gfx1250 (wave32): f16 WMMA + async global->LDS copies
//   B=2, S=2048, D=2048, H=16, dh=128
// ---------------------------------------------------------------------------

typedef __attribute__((ext_vector_type(16))) _Float16 v16h;
typedef __attribute__((ext_vector_type(8)))  _Float16 v8h;
typedef __attribute__((ext_vector_type(8)))  float    v8f;

#define DH   128
#define DHP  (DH + 8)     // padded LDS stride (halves)
#define BKT  32           // GEMM k-tile
#define BKP  (BKT + 8)    // 40 halves = 80B stride (16B multiple)
#define SPP  (32 + 8)     // P-tile stride

static __device__ __forceinline__ v8f wmma_f16(v16h a, v16h b, v8f c) {
  return __builtin_amdgcn_wmma_f32_16x16x32_f16(false, a, false, b, (short)0, c,
                                                false, false);
}

// --- async global->LDS (CDNA5): 16B per lane, tracked by ASYNCcnt ----------
static __device__ __forceinline__ void async_b128(unsigned lds_off,
                                                  const _Float16* g) {
  asm volatile("global_load_async_to_lds_b128 %0, %1, off"
               :
               : "v"(lds_off), "v"(g)
               : "memory");
}
static __device__ __forceinline__ unsigned lds_off(const void* p) {
  return (unsigned)(uintptr_t)p;  // LDS aperture: byte offset in addr[31:0]
}
static __device__ __forceinline__ void wait_async_0() {
#if __has_builtin(__builtin_amdgcn_s_wait_asynccnt)
  __builtin_amdgcn_s_wait_asynccnt(0);
#else
  asm volatile("s_wait_asynccnt 0x0" ::: "memory");
#endif
}
static __device__ __forceinline__ void wait_async_4() {
#if __has_builtin(__builtin_amdgcn_s_wait_asynccnt)
  __builtin_amdgcn_s_wait_asynccnt(4);
#else
  asm volatile("s_wait_asynccnt 0x4" ::: "memory");
#endif
}
static __device__ __forceinline__ void wait_async_8() {
#if __has_builtin(__builtin_amdgcn_s_wait_asynccnt)
  __builtin_amdgcn_s_wait_asynccnt(8);
#else
  asm volatile("s_wait_asynccnt 0x8" ::: "memory");
#endif
}

// --- WMMA fragment loaders (contiguous k = 16*hi + s, same in A and B) -------------------
static __device__ __forceinline__ v16h ldA(const _Float16* base, int ld) {
  const int lane = threadIdx.x & 31;
  const int m = lane & 15, hi = lane >> 4;
  const _Float16* p = base + m * ld;
  v16h f;
#pragma unroll
  for (int i = 0; i < 8; ++i) f[i] = p[16 * hi + i];
#pragma unroll
  for (int i = 0; i < 8; ++i) f[8 + i] = p[16 * hi + 8 + i];
  return f;
}
static __device__ __forceinline__ v16h ldB(const _Float16* base, int ld) {
  const int lane = threadIdx.x & 31;
  const int n = lane & 15, hi = lane >> 4;
  const _Float16* p = base + n * ld + 16 * hi;
  v16h f;
#pragma unroll
  for (int i = 0; i < 16; ++i) f[i] = p[i];
  return f;
}
static __device__ __forceinline__ v16h ldBcol(const _Float16* base, int ld) {
  const int lane = threadIdx.x & 31;
  const int n = lane & 15, hi = lane >> 4;
  v16h f;
#pragma unroll
  for (int i = 0; i < 16; ++i) f[i] = base[(16 * hi + i) * ld + n];
  return f;
}

// ---------------------------------------------------------------------------
// Kernel 0: bulk f32 -> f16 conversion (vectorized 8/thread)
// ---------------------------------------------------------------------------
__global__ __launch_bounds__(256) void cvt_f16(const float* __restrict__ s,
                                               _Float16* __restrict__ d,
                                               int n8) {
  const int i = blockIdx.x * 256 + threadIdx.x;
  if (i >= n8) return;
  const float4* sp = (const float4*)s + (size_t)i * 2;
  const float4 a = sp[0], b = sp[1];
  v8h h;
  h[0] = (_Float16)a.x; h[1] = (_Float16)a.y;
  h[2] = (_Float16)a.z; h[3] = (_Float16)a.w;
  h[4] = (_Float16)b.x; h[5] = (_Float16)b.y;
  h[6] = (_Float16)b.z; h[7] = (_Float16)b.w;
  ((v8h*)d)[i] = h;
}

// ---------------------------------------------------------------------------
// Kernel 1: C[M,N] = A[M,K] * W[N,K]^T, f16 operands, async double-buffered.
// 256 threads (8 waves), tile 128x128, k-step 32, 8 WMMA / wave / k-step.
// ---------------------------------------------------------------------------
template <typename OutT>
__global__ __launch_bounds__(256) void gemm_h(const _Float16* __restrict__ A,
                                              const _Float16* __restrict__ W,
                                              OutT* __restrict__ C, int M,
                                              int N, int K) {
  __shared__ _Float16 sA[2][128][BKP];
  __shared__ _Float16 sB[2][128][BKP];

  const int tid  = threadIdx.x;
  const int lane = tid & 31;
  const int wave = tid >> 5;
  const int wm   = wave & 3;
  const int wn   = wave >> 2;
  const int m0   = blockIdx.y * 128;
  const int n0   = blockIdx.x * 128;
  const int nl   = lane & 15, hi = lane >> 4;

  v8f acc[2][4];
#pragma unroll
  for (int i = 0; i < 2; ++i)
#pragma unroll
    for (int j = 0; j < 4; ++j) acc[i][j] = {};

  const int row = tid >> 1;
  const int c0  = (tid & 1) * 16;  // 16 halves = 32B = 2x b128 per thread

  auto fill = [&](int buf, int k0) {
    const _Float16* ga = A + (size_t)(m0 + row) * K + k0 + c0;
    const _Float16* gb = W + (size_t)(n0 + row) * K + k0 + c0;
    const unsigned la = lds_off(&sA[buf][row][c0]);
    const unsigned lb = lds_off(&sB[buf][row][c0]);
    async_b128(la, ga);
    async_b128(la + 16, ga + 8);
    async_b128(lb, gb);
    async_b128(lb + 16, gb + 8);
  };

  fill(0, 0);
  int cb = 0;
  for (int k0 = 0; k0 < K; k0 += BKT, cb ^= 1) {
    if (k0 + BKT < K) {
      fill(cb ^ 1, k0 + BKT);  // prefetch next tile via async DMA
      wait_async_4();          // oldest 4 (this tile) landed; next stays in flight
    } else {
      wait_async_0();
    }
    __syncthreads();

    v16h af[2], bf[4];
#pragma unroll
    for (int i = 0; i < 2; ++i) af[i] = ldA(&sA[cb][wm * 32 + 16 * i][0], BKP);
#pragma unroll
    for (int j = 0; j < 4; ++j) bf[j] = ldB(&sB[cb][wn * 64 + 16 * j][0], BKP);
#pragma unroll
    for (int i = 0; i < 2; ++i)
#pragma unroll
      for (int j = 0; j < 4; ++j) acc[i][j] = wmma_f16(af[i], bf[j], acc[i][j]);
    __syncthreads();
  }

#pragma unroll
  for (int i = 0; i < 2; ++i)
#pragma unroll
    for (int j = 0; j < 4; ++j)
#pragma unroll
      for (int r = 0; r < 8; ++r) {
        const int gm = m0 + wm * 32 + 16 * i + 8 * hi + r;
        const int gn = n0 + wn * 64 + 16 * j + nl;
        C[(size_t)gm * N + gn] = (OutT)acc[i][j][r];
      }
}

// ---------------------------------------------------------------------------
// Kernel 2: wk[h*128+l] = sum_d W_k_heads[h,l,d]; likewise wv.
// ---------------------------------------------------------------------------
__global__ __launch_bounds__(256) void head_sums(const float* __restrict__ Wk,
                                                 const float* __restrict__ Wv,
                                                 float* __restrict__ wk,
                                                 float* __restrict__ wv) {
  const int e = blockIdx.x * 256 + threadIdx.x;
  if (e < 2048) {
    const float* p = Wk + (size_t)e * 128;
    float a = 0.f;
#pragma unroll 8
    for (int d = 0; d < 128; ++d) a += p[d];
    wk[e] = a;
    p = Wv + (size_t)e * 128;
    float b = 0.f;
#pragma unroll 8
    for (int d = 0; d < 128; ++d) b += p[d];
    wv[e] = b;
  }
}

// ---------------------------------------------------------------------------
// Kernel 3: RoPE + K/V construction. One block per (b,s).
// ---------------------------------------------------------------------------
__global__ __launch_bounds__(256) void rope_build(
    const _Float16* __restrict__ Q1h, const float* __restrict__ kvf32,
    const float* __restrict__ wk, const float* __restrict__ wv,
    _Float16* __restrict__ Qh, _Float16* __restrict__ Kh,
    _Float16* __restrict__ Vh, float* __restrict__ kc_out,
    float* __restrict__ vc_out, int B, int H, int S) {
  __shared__ float qrow[2048];
  __shared__ float kc[128], vc[128];
  __shared__ float cs[32], sn[32];

  const int bs = blockIdx.x;
  const int b = bs / S, s = bs % S;
  const int tid = threadIdx.x;

  const _Float16* qsrc = Q1h + (size_t)bs * 2048;
  for (int i = tid; i < 2048; i += 256) qrow[i] = (float)qsrc[i];
  const float* kvsrc = kvf32 + (size_t)bs * 256;
  if (tid < 128) kc[tid] = kvsrc[tid];
  else vc[tid - 128] = kvsrc[tid];
  if (tid < 32) {
    // freqs = 10000^{-j/32}; ang = s * freq
    const float ang = (float)s * __expf(-(float)tid * 0.28782313662425576f);
    cs[tid] = __cosf(ang);
    sn[tid] = __sinf(ang);
  }
  __syncthreads();

  if (tid < 128) kc_out[(size_t)bs * 128 + tid] = kc[tid];
  else vc_out[(size_t)bs * 128 + (tid - 128)] = vc[tid - 128];

  for (int e = tid; e < 2048; e += 256) {
    const int h = e >> 7, d = e & 127;
    const size_t o = (((size_t)(b * H + h)) * S + s) * 128 + d;
    const float q = qrow[e];
    float qo;
    if (d < 32)      qo = q * cs[d >> 1] - qrow[e + 32] * sn[d >> 1];
    else if (d < 64) qo = q * cs[d >> 1] + qrow[e - 32] * sn[d >> 1];
    else             qo = q;
    Qh[o] = (_Float16)qo;
    const float kd = kc[d] * wk[e];
    float ko;
    if (d < 32)      ko = kd * cs[d >> 1] - (kc[d + 32] * wk[e + 32]) * sn[d >> 1];
    else if (d < 64) ko = kd * cs[d >> 1] + (kc[d - 32] * wk[e - 32]) * sn[d >> 1];
    else             ko = kd;
    Kh[o] = (_Float16)ko;
    Vh[o] = (_Float16)(vc[d] * wv[e]);
  }
}

// ---------------------------------------------------------------------------
// Kernel 4: causal flash attention, async double-buffered K/V tiles.
// grid (S/64, B*H), 128 threads (4 waves); attn emitted f16 (B,S,H*dh).
// ---------------------------------------------------------------------------
__global__ __launch_bounds__(128) void flash_attn(
    const _Float16* __restrict__ Qh, const _Float16* __restrict__ Kh,
    const _Float16* __restrict__ Vh, _Float16* __restrict__ attn, int B, int H,
    int S) {
  __shared__ _Float16 sQ[64][DHP];
  __shared__ _Float16 sK[2][32][DHP];
  __shared__ _Float16 sV[2][32][DHP];
  __shared__ _Float16 sP[4][16][SPP];

  const int bh = blockIdx.y;
  const int b = bh / H, h = bh % H;
  const int q0 = blockIdx.x * 64;
  const int tid = threadIdx.x, wave = tid >> 5, lane = tid & 31;
  const int nl = lane & 15, hi = lane >> 4;
  const float iscale = 0.08838834764831845f;  // 1/sqrt(128)

  const _Float16* Qb = Qh + (size_t)bh * S * DH;
  const _Float16* Kb = Kh + (size_t)bh * S * DH;
  const _Float16* Vb = Vh + (size_t)bh * S * DH;

  {  // Q tile: 64x128, 8 async b128 per thread
    const int row = tid >> 1, c0 = (tid & 1) * 64;
    const _Float16* gq = Qb + (size_t)(q0 + row) * DH + c0;
    const unsigned lq = lds_off(&sQ[row][c0]);
#pragma unroll
    for (int i = 0; i < 8; ++i) async_b128(lq + 16 * i, gq + 8 * i);
  }

  const int rowk = tid >> 2, ck = (tid & 3) * 32;
  auto fillKV = [&](int buf, int k0) {  // 8 async b128 per thread
    const _Float16* gk = Kb + (size_t)(k0 + rowk) * DH + ck;
    const _Float16* gv = Vb + (size_t)(k0 + rowk) * DH + ck;
    const unsigned lk = lds_off(&sK[buf][rowk][ck]);
    const unsigned lv = lds_off(&sV[buf][rowk][ck]);
#pragma unroll
    for (int i = 0; i < 4; ++i) {
      async_b128(lk + 16 * i, gk + 8 * i);
      async_b128(lv + 16 * i, gv + 8 * i);
    }
  };

  v8f o[8];
#pragma unroll
  for (int t = 0; t < 8; ++t) o[t] = {};
  float mrun[8], lrun[8];
#pragma unroll
  for (int r = 0; r < 8; ++r) { mrun[r] = -1e30f; lrun[r] = 0.f; }

  const int qrow0 = q0 + 16 * wave;
  const int nkt = (q0 + 64) >> 5;

  fillKV(0, 0);
  for (int kt = 0; kt < nkt; ++kt) {
    const int k0 = kt << 5;
    const int cb = kt & 1;
    if (kt + 1 < nkt) {
      fillKV(cb ^ 1, k0 + 32);
      wait_async_8();  // current tile (and Q on iter 0) landed
    } else {
      wait_async_0();
    }
    __syncthreads();

    // scores: Q * K^T (two 16x16 tiles over 32 keys)
    v8f s0 = {}, s1 = {};
#pragma unroll
    for (int kk = 0; kk < 4; ++kk) {
      const v16h qa = ldA(&sQ[16 * wave][32 * kk], DHP);
      s0 = wmma_f16(qa, ldB(&sK[cb][0][32 * kk], DHP), s0);
      s1 = wmma_f16(qa, ldB(&sK[cb][16][32 * kk], DHP), s1);
    }

    // mask + online softmax (C-layout: M = r + 8*hi, N = nl)
    float rmax[8], pscale[8], rsum[8];
#pragma unroll
    for (int r = 0; r < 8; ++r) {
      const int qi = qrow0 + 8 * hi + r;
      float a = s0[r] * iscale, c = s1[r] * iscale;
      if (k0 + nl > qi) a = -1e30f;
      if (k0 + 16 + nl > qi) c = -1e30f;
      s0[r] = a; s1[r] = c;
      rmax[r] = fmaxf(a, c);
    }
#pragma unroll
    for (int off = 8; off >= 1; off >>= 1)
#pragma unroll
      for (int r = 0; r < 8; ++r)
        rmax[r] = fmaxf(rmax[r], __shfl_xor(rmax[r], off, 32));
#pragma unroll
    for (int r = 0; r < 8; ++r) {
      const float mn = fmaxf(mrun[r], rmax[r]);
      pscale[r] = __expf(mrun[r] - mn);
      mrun[r] = mn;
      const float p0 = __expf(s0[r] - mn), p1 = __expf(s1[r] - mn);
      s0[r] = p0; s1[r] = p1;
      rsum[r] = p0 + p1;
    }
#pragma unroll
    for (int off = 8; off >= 1; off >>= 1)
#pragma unroll
      for (int r = 0; r < 8; ++r) rsum[r] += __shfl_xor(rsum[r], off, 32);
#pragma unroll
    for (int r = 0; r < 8; ++r) lrun[r] = lrun[r] * pscale[r] + rsum[r];
#pragma unroll
    for (int t = 0; t < 8; ++t)
#pragma unroll
      for (int r = 0; r < 8; ++r) o[t][r] *= pscale[r];

    // P (C-layout) -> LDS -> A-layout, then O += P * V
#pragma unroll
    for (int r = 0; r < 8; ++r) {
      sP[wave][8 * hi + r][nl] = (_Float16)s0[r];
      sP[wave][8 * hi + r][16 + nl] = (_Float16)s1[r];
    }
    const v16h pa = ldA(&sP[wave][0][0], SPP);  // wave-local LDS, in-order
#pragma unroll
    for (int t = 0; t < 8; ++t)
      o[t] = wmma_f16(pa, ldBcol(&sV[cb][0][16 * t], DHP), o[t]);
    __syncthreads();
  }

  // epilogue: normalize + store f16 (B,S,H*dh)
#pragma unroll
  for (int t = 0; t < 8; ++t)
#pragma unroll
    for (int r = 0; r < 8; ++r) {
      const int qi = qrow0 + 8 * hi + r;
      const int d = 16 * t + nl;
      attn[((size_t)b * S + qi) * (size_t)(H * DH) + h * DH + d] =
          (_Float16)(o[t][r] / lrun[r]);
    }
}

// ---------------------------------------------------------------------------
// Launcher
// ---------------------------------------------------------------------------
extern "C" void kernel_launch(void* const* d_in, const int* in_sizes, int n_in,
                              void* d_out, int out_size, void* d_ws,
                              size_t ws_size, hipStream_t stream) {
  const int B = 2, S = 2048, D = 2048, H = 16;
  const size_t BS = (size_t)B * S;   // 4096
  const size_t BSD = BS * D;         // 8,388,608
  const size_t DD = (size_t)D * D;   // 4,194,304

  const float* x    = (const float*)d_in[0];
  const float* W_q  = (const float*)d_in[1];
  const float* W_kv = (const float*)d_in[2];
  const float* W_kh = (const float*)d_in[3];
  const float* W_vh = (const float*)d_in[4];
  const float* W_o  = (const float*)d_in[5];

  // workspace carve-up (f32 region, then f16 region; all 16B-aligned)
  float* kvf32 = (float*)d_ws;                 // BS*256
  float* wk    = kvf32 + BS * 256;             // 2048
  float* wv    = wk + 2048;                    // 2048
  _Float16* xh    = (_Float16*)(wv + 2048);    // BSD
  _Float16* Wqh   = xh + BSD;                  // DD
  _Float16* Wkvh  = Wqh + DD;                  // 256*D
  _Float16* Woh   = Wkvh + (size_t)256 * D;    // DD
  _Float16* Q1h   = Woh + DD;                  // BSD  (B,S,D)
  _Float16* attnh = Q1h + BSD;                 // BSD  (B,S,D)
  _Float16* Qh    = attnh + BSD;               // BSD  (B,H,S,dh)
  _Float16* Kh    = Qh + BSD;                  // BSD
  _Float16* Vh    = Kh + BSD;                  // BSD

  float* out    = (float*)d_out;               // (B,S,D)
  float* kc_out = out + BSD;                   // (B,S,128)
  float* vc_out = kc_out + BS * 128;           // (B,S,128)

  // 0) one-shot f32 -> f16 of all GEMM operands
  cvt_f16<<<dim3((int)(BSD / 8 / 256)), 256, 0, stream>>>(x, xh, (int)(BSD / 8));
  cvt_f16<<<dim3((int)(DD / 8 / 256)), 256, 0, stream>>>(W_q, Wqh, (int)(DD / 8));
  cvt_f16<<<dim3((int)(256 * D / 8 / 256)), 256, 0, stream>>>(
      W_kv, Wkvh, (int)(256 * D / 8));
  cvt_f16<<<dim3((int)(DD / 8 / 256)), 256, 0, stream>>>(W_o, Woh, (int)(DD / 8));

  // 1) Q = x @ W_q^T   (4096 x 2048 x 2048) -> f16
  gemm_h<_Float16><<<dim3(D / 128, (int)(BS / 128)), 256, 0, stream>>>(
      xh, Wqh, Q1h, (int)BS, D, D);
  // 2) kv = x @ W_kv^T (4096 x 256 x 2048) -> f32 (feeds exact k_c/v_c outputs)
  gemm_h<float><<<dim3(2, (int)(BS / 128)), 256, 0, stream>>>(
      xh, Wkvh, kvf32, (int)BS, 256, D);
  // 3) per-(h,l) weight sums
  head_sums<<<dim3(8), 256, 0, stream>>>(W_kh, W_vh, wk, wv);
  // 4) RoPE + K/V build + k_c/v_c outputs
  rope_build<<<dim3((int)BS), 256, 0, stream>>>(Q1h, kvf32, wk, wv, Qh, Kh, Vh,
                                                kc_out, vc_out, B, H, S);
  // 5) causal flash attention -> f16 attn
  flash_attn<<<dim3(S / 64, B * H), 128, 0, stream>>>(Qh, Kh, Vh, attnh, B, H,
                                                      S);
  // 6) out = attn @ W_o^T (4096 x 2048 x 2048) -> f32
  gemm_h<float><<<dim3(D / 128, (int)(BS / 128)), 256, 0, stream>>>(
      attnh, Woh, out, (int)BS, D, D);
}
